// Network_50096498540948
// MI455X (gfx1250) — matrix operand order repaired and checked by reading, not compiled, not run on hardware
//
#include <hip/hip_runtime.h>
#include <hip/hip_bf16.h>
#include <math.h>

typedef __attribute__((ext_vector_type(16))) _Float16 v16h;
typedef __attribute__((ext_vector_type(8)))  _Float16 v8h;
typedef __attribute__((ext_vector_type(8)))  float    v8f;
typedef unsigned int uint32x4 __attribute__((ext_vector_type(4)));
typedef int          int32x8  __attribute__((ext_vector_type(8)));
typedef int          int32x4  __attribute__((ext_vector_type(4)));

#if defined(__gfx1250__) && __has_builtin(__builtin_amdgcn_tensor_load_to_lds) && \
    __has_builtin(__builtin_amdgcn_s_wait_tensorcnt)
#define HAVE_TDM 1
#else
#define HAVE_TDM 0
#endif

#define N_NODES 20000
#define N_EDGES 320000
#define N_GRAPH 16

#define C_S 0.3826834323650898f
#define C_X 0.9238795325112867f
#define INV_SQRT_NEIGH 0.25f
#define INV_SQRT_NODES 0.028284271247461901f
#define INV_SQRT3 0.5773502691896258f
#define INV_SQRT2 0.7071067811865475f
#define SQRT3F    1.7320508075688772f

__device__ __forceinline__ float fast_rcp(float x) {
#if defined(__gfx1250__) && __has_builtin(__builtin_amdgcn_rcpf)
    return __builtin_amdgcn_rcpf(x);
#else
    return 1.0f / x;
#endif
}
__device__ __forceinline__ float siluf(float x) { return x * fast_rcp(1.0f + __expf(-x)); }
__device__ __forceinline__ float sigmf(float x) { return fast_rcp(1.0f + __expf(-x)); }

// ---------------------------------------------------------------------------
// Swizzled ("fragment-major") weight layout:
//   frag index f, lane l (0..31), half j (0..15)  ->  buf[(f*32+l)*16+j]
// B-fragment mapping (16x16x32 f16): col = (l&15), K = ((l&16)?8:0)+(j&7)+((j&8)?16:0)
// so a B fragment load is one contiguous 32B read per lane.
// ---------------------------------------------------------------------------

// TDM copy (multiple of 8 bytes) from global -> LDS, else fallback wave copy.
__device__ __forceinline__ void stage_to_lds(const _Float16* __restrict__ gsrc,
                                             _Float16* __restrict__ ldst,
                                             int halfs, int lane)
{
#if HAVE_TDM
    (void)lane;
    const unsigned long long ga = (unsigned long long)(uintptr_t)gsrc;
    const unsigned int       la = (unsigned int)(uintptr_t)(void*)ldst;
    const unsigned int units = (unsigned int)(halfs * 2) >> 3;   // 8-byte units
    uint32x4 g0 = { 1u,                                          // count=1 (valid)
                    la,                                          // lds_addr
                    (unsigned int)ga,                            // global_addr lo
                    (unsigned int)((ga >> 32) & 0x01FFFFFFu) | 0x80000000u }; // hi | type=2
    int32x8 g1;
    g1[0] = (int)(3u << 16);                   // data_size = 8B
    g1[1] = (int)(units << 16);                // tensor_dim0[15:0]
    g1[2] = (int)((units >> 16) | (1u << 16)); // tensor_dim0 hi | tensor_dim1 = 1
    g1[3] = (int)(units << 16);                // tile_dim0
    g1[4] = 1;                                 // tile_dim1 = 1
    g1[5] = (int)units;                        // tensor_dim0_stride
    g1[6] = (int)(units << 16);                // tensor_dim1_stride lo
    g1[7] = (int)(units >> 16);                // tensor_dim1_stride hi
    int32x4 gz4 = { 0, 0, 0, 0 };
    int32x8 gz8 = { 0, 0, 0, 0, 0, 0, 0, 0 };
    __builtin_amdgcn_tensor_load_to_lds(g0, g1, gz4, gz4, gz8, 0);
#else
    for (int i = lane; i < halfs; i += 32) ldst[i] = gsrc[i];
#endif
}

__device__ __forceinline__ void wait_stage() {
#if HAVE_TDM
    __builtin_amdgcn_s_wait_tensorcnt((short)0);
#endif
}

// Radial MLP for a 16-edge tile:  w = (silu(emb @ W1 / sqrt(10)) @ W2) / sqrt(100)
// embh: E x 16 f16 (cols 10..15 zero); w1f: 7 frags swizzled (staged via TDM);
// w2f: (OUTP/16)*4 frags swizzled (read direct, coalesced). Result: ldsW (16 x OUTP f32).
template<int OUTP>
__device__ __forceinline__ void radial_tile(const _Float16* __restrict__ embh,
                                            const _Float16* __restrict__ w2f,
                                            int e0, int lane,
                                            const _Float16* ldsW1,
                                            _Float16* ldsH, float* ldsW)
{
    const int row   = lane & 15;
    const int koff  = (lane & 16) ? 8 : 0;
    const int mbase = (lane & 16) ? 8 : 0;
    const int ncol  = lane & 15;

    // A fragment for GEMM1 straight from global: 8 contiguous halfs (16B, coalesced),
    // upper 8 halfs map to K=16..31 which are structurally zero.
    v16h a1;
    {
        const v8h lo = *(const v8h*)(embh + (size_t)(e0 + row) * 16 + koff);
#pragma unroll
        for (int j = 0; j < 8; ++j) { a1[j] = lo[j]; a1[8 + j] = (_Float16)0.0f; }
    }
    if (lane == 0 && e0 + 16 < N_EDGES)
        __builtin_prefetch(embh + (size_t)(e0 + 16) * 16, 0, 1);
    // zero hidden pad columns 112..127
    for (int i = lane; i < 16 * 16; i += 32)
        ldsH[(i >> 4) * 128 + 112 + (i & 15)] = (_Float16)0.0f;
    wait_stage();          // W1 fragments resident in LDS
    __syncthreads();

    // GEMM1: hidden = silu(emb @ W1 / sqrt(10)); B frags = 32B contiguous LDS reads
#pragma unroll
    for (int nt = 0; nt < 7; ++nt) {
        v16h b = *(const v16h*)(ldsW1 + (size_t)(nt * 32 + lane) * 16);
        v8f  c = {};
        c = __builtin_amdgcn_wmma_f32_16x16x32_f16(false, a1, false, b, (short)0, c, false, false);
        const int n = nt * 16 + ncol;
#pragma unroll
        for (int r = 0; r < 8; ++r)
            ldsH[(mbase + r) * 128 + n] = (_Float16)siluf(c[r] * 0.31622776601683794f);
    }
    __syncthreads();

    // GEMM2 A fragments (depend only on kt): 2 x ds_load_b128 each, hoisted
    v16h aH[4];
#pragma unroll
    for (int kt = 0; kt < 4; ++kt) {
        const v8h lo = *(const v8h*)(ldsH + row * 128 + kt * 32 + koff);
        const v8h hi = *(const v8h*)(ldsH + row * 128 + kt * 32 + koff + 16);
#pragma unroll
        for (int j = 0; j < 8; ++j) { aH[kt][j] = lo[j]; aH[kt][8 + j] = hi[j]; }
    }

    // GEMM2: w = hidden @ W2 / sqrt(100); B frags = 32B contiguous coalesced global
#pragma unroll
    for (int ot = 0; ot < OUTP / 16; ++ot) {
        v8f c = {};
#pragma unroll
        for (int kt = 0; kt < 4; ++kt) {
            v16h b = *(const v16h*)(w2f + (size_t)((ot * 4 + kt) * 32 + lane) * 16);
            c = __builtin_amdgcn_wmma_f32_16x16x32_f16(false, aH[kt], false, b, (short)0, c, false, false);
        }
        const int n = ot * 16 + ncol;
#pragma unroll
        for (int r = 0; r < 8; ++r)
            ldsW[(mbase + r) * OUTP + n] = c[r] * 0.1f;
    }
    __syncthreads();
}

// ---------------------------------------------------------------------------
// Utility kernels
// ---------------------------------------------------------------------------
__global__ void k_zero_f32(float* p, size_t n) {
    size_t i = (size_t)blockIdx.x * blockDim.x + threadIdx.x;
    for (; i < n; i += (size_t)gridDim.x * blockDim.x) p[i] = 0.0f;
}

__global__ void k_edge_prep(const float* __restrict__ evec,
                            _Float16* __restrict__ embh,
                            float* __restrict__ a0g, float* __restrict__ a1g)
{
    int e = blockIdx.x * blockDim.x + threadIdx.x;
    if (e >= N_EDGES) return;
    float x = evec[e * 3 + 0], y = evec[e * 3 + 1], z = evec[e * 3 + 2];
    float len = sqrtf(x * x + y * y + z * z);
    float inv = fast_rcp(fmaxf(len, 1e-12f));
    float p = len * (1.0f / 2.5f);
    float u = 2.0f * (p - 1.0f);
    float cut;
    if (u > 0.0f)       cut = 0.0f;
    else if (u < -1.0f) cut = 1.0f;
    else                cut = 0.5f * (1.0f - cosf(3.14159265358979323846f * u));
    a0g[e] = cut;
    float s = cut * SQRT3F * inv;
    a1g[e * 3 + 0] = s * x;
    a1g[e * 3 + 1] = s * y;
    a1g[e * 3 + 2] = s * z;
    const float scl  = 3.1622776601683795f / 1.12f;
#pragma unroll
    for (int k = 0; k < 16; ++k) {
        float v = 0.0f;
        if (k < 10) {
            float d = (len - (2.5f / 9.0f) * (float)k) * (9.0f / 2.5f);
            v = __expf(-d * d) * scl;
        }
        embh[(size_t)e * 16 + k] = (_Float16)v;
    }
}

// W1 (10x100 f32) -> 7 swizzled B fragments (3584 halfs)
__global__ void k_swz_w1(const float* __restrict__ w1, _Float16* __restrict__ w1f) {
    int i = blockIdx.x * blockDim.x + threadIdx.x;
    if (i >= 7 * 32 * 16) return;
    int j = i & 15, lane = (i >> 4) & 31, nt = i >> 9;
    int col = nt * 16 + (lane & 15);
    int k   = ((lane & 16) ? 8 : 0) + (j & 7) + ((j & 8) ? 16 : 0);
    float v = (k < 10 && col < 100) ? w1[k * 100 + col] : 0.0f;
    w1f[i] = (_Float16)v;
}

// W2 (100xOUT f32) -> (OUTP/16)*4 swizzled B fragments (OUTP*128 halfs)
__global__ void k_swz_w2(const float* __restrict__ w2, _Float16* __restrict__ w2f,
                         int OUT, int OUTP) {
    int i = blockIdx.x * blockDim.x + threadIdx.x;
    if (i >= OUTP * 128) return;
    int j = i & 15, lane = (i >> 4) & 31, f = i >> 9;  // f = ot*4+kt
    int kt = f & 3, ot = f >> 2;
    int col = ot * 16 + (lane & 15);
    int k   = kt * 32 + ((lane & 16) ? 8 : 0) + (j & 7) + ((j & 8) ? 16 : 0);
    float v = (k < 100 && col < OUT) ? w2[k * OUT + col] : 0.0f;
    w2f[i] = (_Float16)v;
}

// ---------------------------------------------------------------------------
// Fused layer kernels: radial WMMA + messages + atomic scatter
// ---------------------------------------------------------------------------
__global__ __launch_bounds__(32)
void k_layer1_edges(const _Float16* __restrict__ embh,
                    const _Float16* __restrict__ w1f, const _Float16* __restrict__ w2f,
                    const int* __restrict__ src, const int* __restrict__ dst,
                    const float* __restrict__ a0g, const float* __restrict__ a1g,
                    const float* __restrict__ h0p,
                    float* __restrict__ m0, float* __restrict__ m1)
{
    __shared__ __align__(32) _Float16 ldsW1[7 * 32 * 16];
    __shared__ __align__(32) _Float16 ldsH[16 * 128];
    __shared__ float ldsW[16 * 16];
    const int lane = threadIdx.x;
    const int e0 = blockIdx.x * 16;
    stage_to_lds(w1f, ldsW1, 7 * 32 * 16, lane);
    radial_tile<16>(embh, w2f, e0, lane, ldsW1, ldsH, ldsW);
    const float h0v = h0p[0];
    for (int m = 0; m < 16; ++m) {
        int e = e0 + m;
        if (e >= N_EDGES) break;
        int d = dst[e];
        float w0 = ldsW[m * 16 + 0], w1 = ldsW[m * 16 + 1];
        float A0 = a0g[e];
        if (lane == 0)
            atomicAdd(&m0[d], w0 * h0v * A0 * INV_SQRT_NEIGH);
        else if (lane < 4)
            atomicAdd(&m1[d * 3 + (lane - 1)],
                      w1 * h0v * a1g[e * 3 + (lane - 1)] * INV_SQRT_NEIGH);
    }
    (void)src;
}

__global__ __launch_bounds__(32)
void k_layer2_edges(const _Float16* __restrict__ embh,
                    const _Float16* __restrict__ w1f, const _Float16* __restrict__ w2f,
                    const int* __restrict__ src, const int* __restrict__ dst,
                    const float* __restrict__ a0g, const float* __restrict__ a1g,
                    const float* __restrict__ h0, const float* __restrict__ h1,
                    float* __restrict__ m0, float* __restrict__ m1, float* __restrict__ m1e)
{
    __shared__ __align__(32) _Float16 ldsW1[7 * 32 * 16];
    __shared__ __align__(32) _Float16 ldsH[16 * 128];
    __shared__ float ldsW[16 * 160];
    const int lane = threadIdx.x;
    const int e0 = blockIdx.x * 16;
    stage_to_lds(w1f, ldsW1, 7 * 32 * 16, lane);
    radial_tile<160>(embh, w2f, e0, lane, ldsW1, ldsH, ldsW);
    const int u = lane;
    for (int m = 0; m < 16; ++m) {
        int e = e0 + m;
        if (e >= N_EDGES) break;
        int s = src[e], d = dst[e];
        float A0  = a0g[e];
        float A1x = a1g[e * 3 + 0], A1y = a1g[e * 3 + 1], A1z = a1g[e * 3 + 2];
        const float* wr = &ldsW[m * 160];
        float w0 = wr[u], w1 = wr[32 + u], w2 = wr[64 + u], w3 = wr[96 + u], w4 = wr[128 + u];
        float xs0 = h0[(size_t)s * 32 + u];
        const float* p1 = &h1[((size_t)s * 32 + u) * 3];
        float X = p1[0], Y = p1[1], Z = p1[2];
        float dotv = X * A1x + Y * A1y + Z * A1z;
        atomicAdd(&m0[(size_t)d * 64 + u],      w0 * xs0 * A0 * INV_SQRT_NEIGH);
        atomicAdd(&m0[(size_t)d * 64 + 32 + u], w1 * dotv * INV_SQRT3 * INV_SQRT_NEIGH);
        float c01 = w2 * xs0 * INV_SQRT_NEIGH;
        atomicAdd(&m1[((size_t)d * 64 + u) * 3 + 0], c01 * A1x);
        atomicAdd(&m1[((size_t)d * 64 + u) * 3 + 1], c01 * A1y);
        atomicAdd(&m1[((size_t)d * 64 + u) * 3 + 2], c01 * A1z);
        float c10 = w3 * A0 * INV_SQRT_NEIGH;
        atomicAdd(&m1[((size_t)d * 64 + 32 + u) * 3 + 0], c10 * X);
        atomicAdd(&m1[((size_t)d * 64 + 32 + u) * 3 + 1], c10 * Y);
        atomicAdd(&m1[((size_t)d * 64 + 32 + u) * 3 + 2], c10 * Z);
        float c11 = w4 * INV_SQRT2 * INV_SQRT_NEIGH;
        atomicAdd(&m1e[((size_t)d * 32 + u) * 3 + 0], c11 * (Y * A1z - Z * A1y));
        atomicAdd(&m1e[((size_t)d * 32 + u) * 3 + 1], c11 * (Z * A1x - X * A1z));
        atomicAdd(&m1e[((size_t)d * 32 + u) * 3 + 2], c11 * (X * A1y - Y * A1x));
    }
}

__global__ __launch_bounds__(32)
void k_layer3_edges(const _Float16* __restrict__ embh,
                    const _Float16* __restrict__ w1f, const _Float16* __restrict__ w2f,
                    const int* __restrict__ src, const int* __restrict__ dst,
                    const float* __restrict__ a0g, const float* __restrict__ a1g,
                    const float* __restrict__ h0, const float* __restrict__ h1,
                    float* __restrict__ m0)
{
    __shared__ __align__(32) _Float16 ldsW1[7 * 32 * 16];
    __shared__ __align__(32) _Float16 ldsH[16 * 128];
    __shared__ float ldsW[16 * 64];
    const int lane = threadIdx.x;
    const int e0 = blockIdx.x * 16;
    stage_to_lds(w1f, ldsW1, 7 * 32 * 16, lane);
    radial_tile<64>(embh, w2f, e0, lane, ldsW1, ldsH, ldsW);
    const int u = lane;
    for (int m = 0; m < 16; ++m) {
        int e = e0 + m;
        if (e >= N_EDGES) break;
        int s = src[e], d = dst[e];
        float A0  = a0g[e];
        float A1x = a1g[e * 3 + 0], A1y = a1g[e * 3 + 1], A1z = a1g[e * 3 + 2];
        float w0 = ldsW[m * 64 + u], w1 = ldsW[m * 64 + 32 + u];
        float xs0 = h0[(size_t)s * 32 + u];
        const float* p1 = &h1[((size_t)s * 32 + u) * 3];
        float dotv = p1[0] * A1x + p1[1] * A1y + p1[2] * A1z;
        atomicAdd(&m0[(size_t)d * 64 + u],      w0 * xs0 * A0 * INV_SQRT_NEIGH);
        atomicAdd(&m0[(size_t)d * 64 + 32 + u], w1 * dotv * INV_SQRT3 * INV_SQRT_NEIGH);
    }
}

// ---------------------------------------------------------------------------
// Node-wise kernels
// ---------------------------------------------------------------------------
__global__ void k_node_lin0(const float* __restrict__ x, const float* __restrict__ W,
                            float* __restrict__ y, int rows, int cols) {
    int i = blockIdx.x * blockDim.x + threadIdx.x;
    if (i >= N_NODES * cols) return;
    int n = i / cols, v = i % cols;
    float acc = 0.0f;
    for (int u = 0; u < rows; ++u) acc += x[(size_t)n * rows + u] * W[u * cols + v];
    y[i] = acc * rsqrtf((float)rows);
}

__global__ void k_node_lin1(const float* __restrict__ x, const float* __restrict__ W,
                            float* __restrict__ y) {
    int i = blockIdx.x * blockDim.x + threadIdx.x;
    if (i >= N_NODES * 32 * 3) return;
    int d = i % 3, v = (i / 3) % 32, n = i / 96;
    float acc = 0.0f;
    for (int u = 0; u < 32; ++u) acc += x[((size_t)n * 32 + u) * 3 + d] * W[u * 32 + v];
    y[i] = acc * 0.17677669529663687f;
}

__global__ void k_c1_finish(const float* __restrict__ m0, const float* __restrict__ m1,
                            const float* __restrict__ sc0e, const float* __restrict__ l20e,
                            const float* __restrict__ l21o,
                            float* __restrict__ x0, float* __restrict__ x1) {
    int i = blockIdx.x * blockDim.x + threadIdx.x;
    if (i >= N_NODES * 32) return;
    int n = i / 32, u = i % 32;
    float m0v = m0[n];
    float y0a = C_S * sc0e[u]      + C_X * m0v * l20e[u];
    float y0b = C_S * sc0e[32 + u] + C_X * m0v * l20e[32 + u];
    x0[i] = siluf(y0a);
    float g = sigmf(y0b);
#pragma unroll
    for (int d = 0; d < 3; ++d)
        x1[(size_t)i * 3 + d] = m1[n * 3 + d] * l21o[u] * g;
}

__global__ void k_c2_finish(const float* __restrict__ m0, const float* __restrict__ m1,
                            const float* __restrict__ m1e,
                            const float* __restrict__ s0, const float* __restrict__ s1,
                            const float* __restrict__ W0, const float* __restrict__ W1o,
                            const float* __restrict__ W1e,
                            float* __restrict__ x0, float* __restrict__ x1,
                            float* __restrict__ x1e) {
    int i = blockIdx.x * blockDim.x + threadIdx.x;
    if (i >= N_NODES * 32) return;
    int n = i / 32, u = i % 32;
    float a0 = 0.f, a1 = 0.f, a2 = 0.f;
    for (int k = 0; k < 64; ++k) {
        float mv = m0[(size_t)n * 64 + k];
        a0 += mv * W0[k * 96 + u];
        a1 += mv * W0[k * 96 + 32 + u];
        a2 += mv * W0[k * 96 + 64 + u];
    }
    float y0a = C_S * s0[(size_t)n * 96 + u]            + C_X * a0 * 0.125f;
    float g1  = sigmf(C_S * s0[(size_t)n * 96 + 32 + u] + C_X * a1 * 0.125f);
    float g2  = sigmf(C_S * s0[(size_t)n * 96 + 64 + u] + C_X * a2 * 0.125f);
    x0[i] = siluf(y0a);
#pragma unroll
    for (int d = 0; d < 3; ++d) {
        float acc1 = 0.f, acce = 0.f;
        for (int k = 0; k < 64; ++k) acc1 += m1[((size_t)n * 64 + k) * 3 + d] * W1o[k * 32 + u];
        for (int k = 0; k < 32; ++k) acce += m1e[((size_t)n * 32 + k) * 3 + d] * W1e[k * 32 + u];
        float y1  = C_S * s1[((size_t)n * 32 + u) * 3 + d] + C_X * acc1 * 0.125f;
        float y1e = acce * 0.17677669529663687f;
        x1[((size_t)n * 32 + u) * 3 + d]  = y1 * g1;
        x1e[((size_t)n * 32 + u) * 3 + d] = y1e * g2;
    }
}

__global__ void k_c3_final(const float* __restrict__ x0, const float* __restrict__ sc0e,
                           const float* __restrict__ m0, const float* __restrict__ l20e,
                           const int* __restrict__ batch, float* __restrict__ out) {
    int n = blockIdx.x * blockDim.x + threadIdx.x;
    if (n >= N_NODES) return;
    float s0a = 0.f;
    for (int u = 0; u < 32; ++u) s0a += x0[(size_t)n * 32 + u] * sc0e[u];
    s0a *= 0.17677669529663687f;
    float ma = 0.f;
    for (int k = 0; k < 64; ++k) ma += m0[(size_t)n * 64 + k] * l20e[k];
    ma *= 0.125f;
    atomicAdd(&out[batch[n]], (C_S * s0a + C_X * ma) * INV_SQRT_NODES);
}

// ---------------------------------------------------------------------------
// Host launcher
// ---------------------------------------------------------------------------
extern "C" void kernel_launch(void* const* d_in, const int* in_sizes, int n_in,
                              void* d_out, int out_size, void* d_ws, size_t ws_size,
                              hipStream_t stream) {
    (void)in_sizes; (void)n_in; (void)out_size; (void)ws_size;
    const int*   edge_index = (const int*)d_in[1];
    const float* edge_vec   = (const float*)d_in[2];
    const int*   batch      = (const int*)d_in[3];
    int i = 4;
    const float* c1_sc0e  = (const float*)d_in[i++];
    const float* c1_l10e  = (const float*)d_in[i++];
    const float* c1_fw1   = (const float*)d_in[i++];
    const float* c1_fw2   = (const float*)d_in[i++];
    const float* c1_l20e  = (const float*)d_in[i++];
    const float* c1_l21o  = (const float*)d_in[i++];
    const float* c2_sc0e  = (const float*)d_in[i++];
    const float* c2_sc1o  = (const float*)d_in[i++];
    const float* c2_l10e  = (const float*)d_in[i++];
    const float* c2_l11o  = (const float*)d_in[i++];
    const float* c2_fw1   = (const float*)d_in[i++];
    const float* c2_fw2   = (const float*)d_in[i++];
    const float* c2_l20e  = (const float*)d_in[i++];
    const float* c2_l21o  = (const float*)d_in[i++];
    const float* c2_l21e  = (const float*)d_in[i++];
    const float* c3_sc0e  = (const float*)d_in[i++];
    const float* c3_l10e  = (const float*)d_in[i++];
    const float* c3_l11o  = (const float*)d_in[i++];
    i++; // c3_lin1_1e : computed but unused in reference
    const float* c3_fw1   = (const float*)d_in[i++];
    const float* c3_fw2   = (const float*)d_in[i++];
    const float* c3_l20e  = (const float*)d_in[i++];

    const int* src = edge_index;
    const int* dst = edge_index + N_EDGES;

    char* ws = (char*)d_ws;
    auto alloc = [&](size_t bytes) -> char* {
        char* p = ws; ws += (bytes + 255) & ~(size_t)255; return p;
    };
    _Float16* embh = (_Float16*)alloc((size_t)N_EDGES * 16 * 2);
    float*    a0g  = (float*)alloc((size_t)N_EDGES * 4);
    float*    a1g  = (float*)alloc((size_t)N_EDGES * 12);
    _Float16* w1f  = (_Float16*)alloc(7 * 32 * 16 * 2);
    _Float16* w2f  = (_Float16*)alloc((size_t)160 * 128 * 2);
    float* x0  = (float*)alloc((size_t)N_NODES * 32 * 4);
    float* x1  = (float*)alloc((size_t)N_NODES * 96 * 4);
    float* x1e = (float*)alloc((size_t)N_NODES * 96 * 4);
    float* h0b = (float*)alloc((size_t)N_NODES * 32 * 4);
    float* h1b = (float*)alloc((size_t)N_NODES * 96 * 4);
    float* s0b = (float*)alloc((size_t)N_NODES * 96 * 4);
    float* s1b = (float*)alloc((size_t)N_NODES * 96 * 4);
    float* m0b = (float*)alloc((size_t)N_NODES * 64 * 4);
    float* m1b = (float*)alloc((size_t)N_NODES * 192 * 4);
    float* m1eb= (float*)alloc((size_t)N_NODES * 96 * 4);

    const int edgeBlocks = (N_EDGES + 15) / 16;
    const int T = 256;
    auto grid = [](size_t n, int t) { return (int)((n + t - 1) / t); };

    k_zero_f32<<<1, 64, 0, stream>>>((float*)d_out, (size_t)N_GRAPH);
    k_edge_prep<<<grid(N_EDGES, T), T, 0, stream>>>(edge_vec, embh, a0g, a1g);

    // ---- layer 1 ----
    k_swz_w1<<<grid(7 * 32 * 16, T), T, 0, stream>>>(c1_fw1, w1f);
    k_swz_w2<<<grid(16 * 128, T), T, 0, stream>>>(c1_fw2, w2f, 2, 16);
    k_zero_f32<<<grid(N_NODES, T), T, 0, stream>>>(m0b, (size_t)N_NODES);
    k_zero_f32<<<grid((size_t)N_NODES * 3, T), T, 0, stream>>>(m1b, (size_t)N_NODES * 3);
    k_layer1_edges<<<edgeBlocks, 32, 0, stream>>>(embh, w1f, w2f, src, dst, a0g, a1g,
                                                  c1_l10e, m0b, m1b);
    k_c1_finish<<<grid((size_t)N_NODES * 32, T), T, 0, stream>>>(m0b, m1b, c1_sc0e,
                                                                 c1_l20e, c1_l21o, x0, x1);

    // ---- layer 2 ----
    k_node_lin0<<<grid((size_t)N_NODES * 32, T), T, 0, stream>>>(x0, c2_l10e, h0b, 32, 32);
    k_node_lin1<<<grid((size_t)N_NODES * 96, T), T, 0, stream>>>(x1, c2_l11o, h1b);
    k_node_lin0<<<grid((size_t)N_NODES * 96, T), T, 0, stream>>>(x0, c2_sc0e, s0b, 32, 96);
    k_node_lin1<<<grid((size_t)N_NODES * 96, T), T, 0, stream>>>(x1, c2_sc1o, s1b);
    k_swz_w1<<<grid(7 * 32 * 16, T), T, 0, stream>>>(c2_fw1, w1f);
    k_swz_w2<<<grid(160 * 128, T), T, 0, stream>>>(c2_fw2, w2f, 160, 160);
    k_zero_f32<<<grid((size_t)N_NODES * 64, T), T, 0, stream>>>(m0b, (size_t)N_NODES * 64);
    k_zero_f32<<<grid((size_t)N_NODES * 192, T), T, 0, stream>>>(m1b, (size_t)N_NODES * 192);
    k_zero_f32<<<grid((size_t)N_NODES * 96, T), T, 0, stream>>>(m1eb, (size_t)N_NODES * 96);
    k_layer2_edges<<<edgeBlocks, 32, 0, stream>>>(embh, w1f, w2f, src, dst, a0g, a1g,
                                                  h0b, h1b, m0b, m1b, m1eb);
    k_c2_finish<<<grid((size_t)N_NODES * 32, T), T, 0, stream>>>(m0b, m1b, m1eb, s0b, s1b,
                                                                 c2_l20e, c2_l21o, c2_l21e,
                                                                 x0, x1, x1e);

    // ---- layer 3 ----
    k_node_lin0<<<grid((size_t)N_NODES * 32, T), T, 0, stream>>>(x0, c3_l10e, h0b, 32, 32);
    k_node_lin1<<<grid((size_t)N_NODES * 96, T), T, 0, stream>>>(x1, c3_l11o, h1b);
    k_swz_w1<<<grid(7 * 32 * 16, T), T, 0, stream>>>(c3_fw1, w1f);
    k_swz_w2<<<grid(64 * 128, T), T, 0, stream>>>(c3_fw2, w2f, 64, 64);
    k_zero_f32<<<grid((size_t)N_NODES * 64, T), T, 0, stream>>>(m0b, (size_t)N_NODES * 64);
    k_layer3_edges<<<edgeBlocks, 32, 0, stream>>>(embh, w1f, w2f, src, dst, a0g, a1g,
                                                  h0b, h1b, m0b);
    k_c3_final<<<grid(N_NODES, T), T, 0, stream>>>(x0, c3_sc0e, m0b, c3_l20e, batch,
                                                   (float*)d_out);
    (void)x1e;
}
